// QuantizedLinear_77859167142257
// MI455X (gfx1250) — 8-bit matrix operand order repaired and checked by reading, compiled, not run on hardware
//
#include <hip/hip_runtime.h>
#include <stdint.h>

typedef __attribute__((ext_vector_type(8))) int   v8i;
typedef __attribute__((ext_vector_type(4))) int   v4i;
typedef __attribute__((ext_vector_type(2))) int   v2i;
typedef __attribute__((ext_vector_type(4))) float v4f;

#define IN_F   2048
#define OUT_F  8192
#define MROWS  8192
#define BM     128
#define BN     128
#define BK     64
#define LDSTR  80
#define ABYTES (BM * LDSTR)
#define BUFBYTES (ABYTES + BN * LDSTR)
#define KSTEPS (IN_F / BK)
#define STAGES 4

template <int N>
static __device__ __forceinline__ void ql_wait_async() {
#if __has_builtin(__builtin_amdgcn_s_wait_asynccnt)
    __builtin_amdgcn_s_wait_asynccnt(N);
#else
    asm volatile("s_wait_asynccnt %0" :: "i"(N) : "memory");
#endif
}

static __device__ __forceinline__ void ql_async_copy32(unsigned lds_off, unsigned& voff,
                                                       const signed char* sbase) {
    asm volatile("global_load_async_to_lds_b128 %[lds], %[voff], %[base]\n\t"
                 "global_load_async_to_lds_b128 %[lds], %[voff], %[base] offset:16\n\t"
                 "v_add_nc_u32 %[voff], 64, %[voff]"
                 : [voff] "+v"(voff)
                 : [lds] "v"(lds_off), [base] "s"(sbase)
                 : "memory");
}

__global__ __launch_bounds__(256) void ql_absmax_kernel(const v4f* __restrict__ x,
                                                        unsigned* __restrict__ dst, int n4) {
    float m = 0.0f;
    for (int i = blockIdx.x * 256 + threadIdx.x; i < n4; i += gridDim.x * 256) {
        v4f v = x[i];
        m = fmaxf(m, fmaxf(fmaxf(fabsf(v[0]), fabsf(v[1])),
                           fmaxf(fabsf(v[2]), fabsf(v[3]))));
    }
    __shared__ float red[256];
    red[threadIdx.x] = m;
    __syncthreads();
    for (int s = 128; s > 0; s >>= 1) {
        if (threadIdx.x < s) red[threadIdx.x] = fmaxf(red[threadIdx.x], red[threadIdx.x + s]);
        __syncthreads();
    }
    if (threadIdx.x == 0) atomicMax(dst, __float_as_uint(red[0]));
}

__global__ void ql_finalize_scale_kernel(unsigned* __restrict__ wsu) {
    float am = __uint_as_float(wsu[0]);
    float s  = (am == 0.0f) ? 1.0f : am * (1.0f / 127.0f);
    ((float*)wsu)[1] = s;
    ((float*)wsu)[2] = 1.0f / s;
}

__global__ __launch_bounds__(256) void ql_quantize_x_kernel(const v4f* __restrict__ x,
                                                            int* __restrict__ xq,
                                                            const float* __restrict__ wsf,
                                                            int n4) {
    int i = blockIdx.x * 256 + threadIdx.x;
    if (i >= n4) return;
    const float inv = wsf[2];
    v4f v = x[i];
    int q0 = (int)fminf(fmaxf(rintf(v[0] * inv), -127.0f), 127.0f);
    int q1 = (int)fminf(fmaxf(rintf(v[1] * inv), -127.0f), 127.0f);
    int q2 = (int)fminf(fmaxf(rintf(v[2] * inv), -127.0f), 127.0f);
    int q3 = (int)fminf(fmaxf(rintf(v[3] * inv), -127.0f), 127.0f);
    xq[i] = (q0 & 0xff) | ((q1 & 0xff) << 8) | ((q2 & 0xff) << 16) | ((q3 & 0xff) << 24);
}

__global__ __launch_bounds__(256) void ql_convert_w_kernel(const v4f* __restrict__ w,
                                                           int* __restrict__ wq, int n4) {
    int i = blockIdx.x * 256 + threadIdx.x;
    if (i >= n4) return;
    v4f v = w[i];
    int q0 = (int)fminf(fmaxf(rintf(v[0]), -127.0f), 127.0f);
    int q1 = (int)fminf(fmaxf(rintf(v[1]), -127.0f), 127.0f);
    int q2 = (int)fminf(fmaxf(rintf(v[2]), -127.0f), 127.0f);
    int q3 = (int)fminf(fmaxf(rintf(v[3]), -127.0f), 127.0f);
    wq[i] = (q0 & 0xff) | ((q1 & 0xff) << 8) | ((q2 & 0xff) << 16) | ((q3 & 0xff) << 24);
}

__global__ __launch_bounds__(256) void ql_gemm_iu8_kernel(
        const signed char* __restrict__ xq, const signed char* __restrict__ wq,
        const float* __restrict__ wscale, const float* __restrict__ bias,
        const float* __restrict__ wsf, float* __restrict__ out) {
    __shared__ __align__(16) signed char lds[STAGES][BUFBYTES];

    const int tid  = threadIdx.x;
    const int bm   = blockIdx.y, bn = blockIdx.x;
    const int wid  = tid >> 5, lane = tid & 31;
    const int wm   = wid >> 2;
    const int wn   = wid & 3;
    const int ln   = lane & 15, lh = lane >> 4;

    v8i acc[4][2];
    #pragma unroll
    for (int mt = 0; mt < 4; ++mt)
        #pragma unroll
        for (int nt = 0; nt < 2; ++nt)
            acc[mt][nt] = {};

    const int r = tid >> 1;
    const int c = (tid & 1) * 32;
    const signed char* baseA = xq + (size_t)bm * BM * IN_F;
    const signed char* baseB = wq + (size_t)bn * BN * IN_F;
    unsigned voffA = (unsigned)(r * IN_F + c);
    unsigned voffB = (unsigned)(r * IN_F + c);
    unsigned aoff[STAGES], boff[STAGES];
    #pragma unroll
    for (int b = 0; b < STAGES; ++b) {
        aoff[b] = (unsigned)(uintptr_t)&lds[b][r * LDSTR + c];
        boff[b] = (unsigned)(uintptr_t)&lds[b][ABYTES + r * LDSTR + c];
    }

    auto issue = [&](int buf) {
        ql_async_copy32(aoff[buf], voffA, baseA);
        ql_async_copy32(boff[buf], voffB, baseB);
    };

    auto compute = [&](int buf) {
        const signed char* ldsA = &lds[buf][0];
        const signed char* ldsB = &lds[buf][ABYTES];
        v8i bf[2];
        #pragma unroll
        for (int nt = 0; nt < 2; ++nt) {
            const signed char* p = ldsB + (wn * 32 + nt * 16 + ln) * LDSTR + lh * 16;
            v4i lo = *(const v4i*)p;
            v4i hi = *(const v4i*)(p + 32);
            bf[nt][0] = lo[0]; bf[nt][1] = lo[1]; bf[nt][2] = lo[2]; bf[nt][3] = lo[3];
            bf[nt][4] = hi[0]; bf[nt][5] = hi[1]; bf[nt][6] = hi[2]; bf[nt][7] = hi[3];
        }
        #pragma unroll
        for (int mt = 0; mt < 4; ++mt) {
            const signed char* q = ldsA + (wm * 64 + mt * 16 + ln) * LDSTR + lh * 16;
            v8i af;
            #pragma unroll
            for (int g = 0; g < 4; ++g) {
                v2i p2 = *(const v2i*)(q + (g & 1) * 8 + (g >> 1) * 32);
                af[2 * g]     = p2[0];
                af[2 * g + 1] = p2[1];
            }
            #pragma unroll
            for (int nt = 0; nt < 2; ++nt)
                acc[mt][nt] = __builtin_amdgcn_wmma_i32_16x16x64_iu8(
                     true, af,  true, bf[nt],
                    acc[mt][nt],  false,  false);
        }
    };

    issue(0);
    issue(1);
    issue(2);

    #pragma unroll
    for (int s = 0; s < KSTEPS - 2; ++s) {
        ql_wait_async<2 * 4>();
        __syncthreads();
        issue((s + 3) & (STAGES - 1));
        compute(s & (STAGES - 1));
    }
    ql_wait_async<1 * 4>();
    __syncthreads();
    compute((KSTEPS - 2) & (STAGES - 1));
    ql_wait_async<0>();
    __syncthreads();
    compute((KSTEPS - 1) & (STAGES - 1));

    const float sx = wsf[1];
    #pragma unroll
    for (int nt = 0; nt < 2; ++nt) {
        const int gn  = bn * BN + wn * 32 + nt * 16 + ln;
        const float f = sx * wscale[gn];
        const float b = bias[gn];
        #pragma unroll
        for (int mt = 0; mt < 4; ++mt) {
            const int gm = bm * BM + wm * 64 + mt * 16 + lh * 8;
            float* o = out + (size_t)gm * OUT_F + gn;
            #pragma unroll
            for (int rr = 0; rr < 8; ++rr)
                o[(size_t)rr * OUT_F] = (float)acc[mt][nt][rr] * f + b;
        }
    }
}

extern "C" void kernel_launch(void* const* d_in, const int* in_sizes, int n_in,
                              void* d_out, int out_size, void* d_ws, size_t ws_size,
                              hipStream_t stream) {
    (void)in_sizes; (void)n_in; (void)out_size; (void)ws_size;
    const float* x     = (const float*)d_in[0];
    const float* w     = (const float*)d_in[1];
    const float* wscal = (const float*)d_in[2];
    const float* bias  = (const float*)d_in[3];
    float* out         = (float*)d_out;

    unsigned char* wsp = (unsigned char*)d_ws;
    unsigned* absbits  = (unsigned*)wsp;
    float* scales      = (float*)wsp;
    signed char* xq    = (signed char*)(wsp + 64);
    signed char* wq    = (signed char*)(wsp + 64 + (size_t)MROWS * IN_F);

    const int n4x = MROWS * IN_F / 4;
    const int n4w = OUT_F * IN_F / 4;

    hipMemsetAsync(d_ws, 0, 16, stream);
    ql_absmax_kernel<<<1024, 256, 0, stream>>>((const v4f*)x, absbits, n4x);
    ql_finalize_scale_kernel<<<1, 1, 0, stream>>>(absbits);
    ql_quantize_x_kernel<<<n4x / 256, 256, 0, stream>>>((const v4f*)x, (int*)xq, scales, n4x);
    ql_convert_w_kernel<<<n4w / 256, 256, 0, stream>>>((const v4f*)w, (int*)wq, n4w);

    dim3 grid(OUT_F / BN, MROWS / BM);
    ql_gemm_iu8_kernel<<<grid, 256, 0, stream>>>(xq, wq, wscal, bias, scales, out);
}
